// VaswaniMultiHeadAttention_50354196578986
// MI455X (gfx1250) — matrix operand order repaired and checked by reading, not compiled, not run on hardware
//
#include <hip/hip_runtime.h>
#include <hip/hip_bf16.h>
#include <stdint.h>

// MI455X / gfx1250, wave32. Matrix math on v_wmma_f32_16x16x32_f16.
// Round 2: double-buffered GEMM pipeline, async global->LDS (ASYNCcnt) for the
// attention K tile, b64 V loads, 8-wave attention blocks, global_prefetch.
//
// Scratch layout in d_ws (needs 3 * B*S*D * 2 bytes = 48 MB):
//   [Qp f16 | Kp f16 | Vp f16]

typedef __attribute__((ext_vector_type(16))) _Float16 v16h;
typedef __attribute__((ext_vector_type(8)))  float    v8f;

#define B_  4
#define S_  2048
#define D_  1024
#define H_  16
#define DH_ 64

union Frag16 { v16h v; unsigned int u[8]; };
union U2H    { uint2 u; _Float16 h[4]; };

__device__ __forceinline__ v8f wmma_f16(const v16h& a, const v16h& b, const v8f& c) {
  // D = A(16x32 f16) * B(32x16 f16) + C(16x16 f32)
  return __builtin_amdgcn_wmma_f32_16x16x32_f16(false, a, false, b, (short)0, c, false, false);
}

// CDNA5 async copy: 16 bytes global -> LDS per lane, tracked by ASYNCcnt.
__device__ __forceinline__ void async_copy_b128(unsigned long long gaddr, unsigned lds_off) {
  asm volatile("global_load_async_to_lds_b128 %0, %1, off"
               :: "v"(lds_off), "v"(gaddr) : "memory");
}
__device__ __forceinline__ void async_wait0() {
  asm volatile("s_wait_asynccnt 0x0" ::: "memory");
}

// ---------------------------------------------------------------------------
// Projection GEMM: C[M=B*S, N=D] = A[M,K=D] * W[K,N] + bias, output f16.
// Block 256 threads (8 waves), tile 128x128, K-step 32, double-buffered LDS.
// ---------------------------------------------------------------------------
__global__ __launch_bounds__(256)
void proj_gemm_kernel(const float* __restrict__ A, const float* __restrict__ W,
                      const float* __restrict__ bias, _Float16* __restrict__ out16)
{
  // As: [m][k] row-major (K pairs contiguous -> 32-bit ds loads for A frags)
  // Bs: [n][k] n-major   (K pairs contiguous -> 32-bit ds loads for B frags)
  __shared__ _Float16 As[2][128][36];
  __shared__ _Float16 Bs[2][128][36];

  const int t    = threadIdx.x;
  const int m0   = blockIdx.x * 128;
  const int n0   = blockIdx.y * 128;
  const int w    = t >> 5;
  const int lane = t & 31;
  const int wm   = w >> 2;        // 0..1  -> 64 rows
  const int wn   = w & 3;         // 0..3  -> 32 cols
  const int hl   = lane >> 4;     // lane half (0/1)
  const int lp   = lane & 15;

  const int ar  = t >> 1, acb = (t & 1) * 16;   // A tile: 2 thr/row, 16 floats each
  const int bkr = t >> 3, bcb = (t & 7) * 16;   // W tile: 8 thr/row, 16 floats each
  const float* Abase = A + (size_t)(m0 + ar) * D_ + acb;
  const float* Wbase = W + (size_t)bkr * D_ + n0 + bcb;

  v8f acc[4][2];
  #pragma unroll
  for (int i = 0; i < 4; ++i)
    #pragma unroll
    for (int j = 0; j < 2; ++j)
      #pragma unroll
      for (int r = 0; r < 8; ++r) acc[i][j][r] = 0.f;

  float4 ra[4], rb[4];

  auto fetch = [&](int k0) {
    const float4* pa = reinterpret_cast<const float4*>(Abase + k0);
    const float4* pb = reinterpret_cast<const float4*>(Wbase + (size_t)k0 * D_);
    #pragma unroll
    for (int i = 0; i < 4; ++i) { ra[i] = pa[i]; rb[i] = pb[i]; }
    if (k0 + 32 < D_) {           // one tile ahead -> global_prefetch_b8
      __builtin_prefetch(Abase + k0 + 32, 0, 0);
      __builtin_prefetch(Wbase + (size_t)(k0 + 32) * D_, 0, 0);
    }
  };
  auto stage = [&](int buf) {
    #pragma unroll
    for (int i = 0; i < 4; ++i) {
      As[buf][ar][acb + 4*i + 0] = (_Float16)ra[i].x;
      As[buf][ar][acb + 4*i + 1] = (_Float16)ra[i].y;
      As[buf][ar][acb + 4*i + 2] = (_Float16)ra[i].z;
      As[buf][ar][acb + 4*i + 3] = (_Float16)ra[i].w;
      Bs[buf][bcb + 4*i + 0][bkr] = (_Float16)rb[i].x;
      Bs[buf][bcb + 4*i + 1][bkr] = (_Float16)rb[i].y;
      Bs[buf][bcb + 4*i + 2][bkr] = (_Float16)rb[i].z;
      Bs[buf][bcb + 4*i + 3][bkr] = (_Float16)rb[i].w;
    }
  };
  auto compute = [&](int buf) {
    Frag16 bf[2];
    #pragma unroll
    for (int nt = 0; nt < 2; ++nt) {
      const int n = wn * 32 + nt * 16 + lp;
      #pragma unroll
      for (int j = 0; j < 8; ++j) {
        const int kk = ((j < 4) ? 0 : 16) + hl * 8 + 2 * (j & 3);  // B: K = base + hl*8 + 2*(j&3)
        bf[nt].u[j] = *reinterpret_cast<const unsigned int*>(&Bs[buf][n][kk]);
      }
    }
    #pragma unroll
    for (int mt = 0; mt < 4; ++mt) {
      const int m = wm * 64 + mt * 16 + lp;
      Frag16 af;
      #pragma unroll
      for (int j = 0; j < 8; ++j) {                     // A: K = base + hl*8 + 2*(j&3)
        const int kk = ((j < 4) ? 0 : 16) + hl * 8 + 2 * (j & 3);
        af.u[j] = *reinterpret_cast<const unsigned int*>(&As[buf][m][kk]);
      }
      acc[mt][0] = wmma_f16(af.v, bf[0].v, acc[mt][0]);
      acc[mt][1] = wmma_f16(af.v, bf[1].v, acc[mt][1]);
    }
  };

  // software pipeline: fetch(i+1) | compute(i) | stage(i+1) | barrier
  fetch(0); stage(0); __syncthreads();
  #pragma unroll 1
  for (int ti = 1; ti < D_ / 32; ++ti) {
    fetch(ti * 32);
    compute((ti - 1) & 1);
    stage(ti & 1);
    __syncthreads();
  }
  compute((D_ / 32 - 1) & 1);

  // epilogue: C layout lane=(hl,n=lp), VGPR r -> row r + 8*hl
  #pragma unroll
  for (int nt = 0; nt < 2; ++nt) {
    const int n = n0 + wn * 32 + nt * 16 + lp;
    const float bv = bias[n];
    #pragma unroll
    for (int mt = 0; mt < 4; ++mt) {
      #pragma unroll
      for (int r = 0; r < 8; ++r) {
        const int m = m0 + wm * 64 + mt * 16 + r + 8 * hl;
        out16[(size_t)m * D_ + n] = (_Float16)(acc[mt][nt][r] + bv);
      }
    }
  }
}

// ---------------------------------------------------------------------------
// Flash attention: block = 256 threads (8 waves), grid (S/128, H, B).
// Wave w owns q rows [qb + 16w, qb + 16w + 16). Key blocks of 32.
// K tile staged via CDNA5 async global->LDS; V staged + transposed via VGPRs.
// ---------------------------------------------------------------------------
__global__ __launch_bounds__(256)
void attn_kernel(const _Float16* __restrict__ qp, const _Float16* __restrict__ kp,
                 const _Float16* __restrict__ vp, float* __restrict__ out)
{
  __shared__ _Float16 Ks[32][72];      // [key][dh], 144B row stride (16B aligned)
  __shared__ _Float16 Vs[64][36];      // [dh][key]
  __shared__ _Float16 Ps[8][16][36];   // per-wave P scratch [qrow][key]

  const int t    = threadIdx.x;
  const int w    = t >> 5;
  const int lane = t & 31;
  const int hl   = lane >> 4;
  const int lp   = lane & 15;
  const int qb   = blockIdx.x * 128;
  const int h    = blockIdx.y;
  const int b    = blockIdx.z;

  // Q resident in registers: two A-fragments (dh chunks 0..31, 32..63)
  Frag16 qf[2];
  {
    const size_t qbase = ((size_t)b * S_ + qb + w * 16 + lp) * D_ + (size_t)h * DH_;
    #pragma unroll
    for (int c = 0; c < 2; ++c)
      #pragma unroll
      for (int j = 0; j < 8; ++j) {
        const int kk = c * 32 + ((j < 4) ? 0 : 16) + hl * 8 + 2 * (j & 3);
        qf[c].u[j] = *reinterpret_cast<const unsigned int*>(qp + qbase + kk);
      }
  }

  float m_r[8], l_r[8];
  #pragma unroll
  for (int r = 0; r < 8; ++r) { m_r[r] = -3.0e38f; l_r[r] = 0.f; }
  v8f oacc[4];
  #pragma unroll
  for (int nc = 0; nc < 4; ++nc)
    #pragma unroll
    for (int r = 0; r < 8; ++r) oacc[nc][r] = 0.f;

  const float inv_sqrt_d = 0.03125f;   // 1/sqrt(1024)

  // cooperative tile indices: 8 threads per key row, 8 halfs (16B) each
  const int kr = t >> 3;
  const int cs = (t & 7) * 8;

  for (int kb = 0; kb < S_; kb += 32) {
    const size_t g = ((size_t)b * S_ + kb + kr) * D_ + (size_t)h * DH_ + cs;

    // K tile: one async b128 per lane, no VGPR round-trip (ASYNCcnt)
    async_copy_b128((unsigned long long)(uintptr_t)(kp + g),
                    (unsigned)(uintptr_t)&Ks[kr][cs]);

    // V tile: 2x b64 global loads, transpose-scatter into Vs[dh][key]
    {
      const uint2* pv = reinterpret_cast<const uint2*>(vp + g);
      U2H r0, r1; r0.u = pv[0]; r1.u = pv[1];
      #pragma unroll
      for (int i = 0; i < 4; ++i) {
        Vs[cs + i][kr]     = r0.h[i];
        Vs[cs + 4 + i][kr] = r1.h[i];
      }
    }
    async_wait0();
    __syncthreads();

    // scores: S[16q x 32k] = Q(16x64) * K^T, as two 16x16 tiles
    Frag16 kf[2][2];  // [key-group g][dh-chunk c]
    #pragma unroll
    for (int gg = 0; gg < 2; ++gg)
      #pragma unroll
      for (int c = 0; c < 2; ++c)
        #pragma unroll
        for (int j = 0; j < 8; ++j) {
          const int kk = c * 32 + ((j < 4) ? 0 : 16) + hl * 8 + 2 * (j & 3);      // dh index pair
          kf[gg][c].u[j] = *reinterpret_cast<const unsigned int*>(&Ks[gg * 16 + lp][kk]);
        }

    v8f s0, s1;
    #pragma unroll
    for (int r = 0; r < 8; ++r) { s0[r] = 0.f; s1[r] = 0.f; }
    s0 = wmma_f16(qf[0].v, kf[0][0].v, s0);
    s0 = wmma_f16(qf[1].v, kf[0][1].v, s0);
    s1 = wmma_f16(qf[0].v, kf[1][0].v, s1);
    s1 = wmma_f16(qf[1].v, kf[1][1].v, s1);

    // online softmax over this 32-key block
    float alpha_r[8];
    #pragma unroll
    for (int r = 0; r < 8; ++r) {
      const float a = s0[r] * inv_sqrt_d;
      const float c = s1[r] * inv_sqrt_d;
      float mx = fmaxf(a, c);
      #pragma unroll
      for (int o = 8; o >= 1; o >>= 1) mx = fmaxf(mx, __shfl_xor(mx, o));
      const float mn = fmaxf(m_r[r], mx);
      const float al = __expf(m_r[r] - mn);
      const float p0 = __expf(a - mn);
      const float p1 = __expf(c - mn);
      float rs = p0 + p1;
      #pragma unroll
      for (int o = 8; o >= 1; o >>= 1) rs += __shfl_xor(rs, o);
      l_r[r] = l_r[r] * al + rs;
      m_r[r] = mn;
      alpha_r[r] = al;
      s0[r] = p0;
      s1[r] = p1;
    }
    #pragma unroll
    for (int nc = 0; nc < 4; ++nc)
      #pragma unroll
      for (int r = 0; r < 8; ++r) oacc[nc][r] *= alpha_r[r];

    // C-layout -> A-layout for P via per-wave LDS (same-wave DS is in-order)
    #pragma unroll
    for (int r = 0; r < 8; ++r) {
      Ps[w][r + 8 * hl][lp]      = (_Float16)s0[r];
      Ps[w][r + 8 * hl][16 + lp] = (_Float16)s1[r];
    }
    Frag16 pf;
    #pragma unroll
    for (int j = 0; j < 8; ++j) {
      const int kk = ((j < 4) ? 0 : 16) + hl * 8 + 2 * (j & 3);
      pf.u[j] = *reinterpret_cast<const unsigned int*>(&Ps[w][lp][kk]);
    }

    // O += P(16x32) * V(32x64): 4 WMMAs over dh chunks of 16
    #pragma unroll
    for (int nc = 0; nc < 4; ++nc) {
      Frag16 vf;
      #pragma unroll
      for (int j = 0; j < 8; ++j) {
        const int kk = ((j < 4) ? 0 : 16) + hl * 8 + 2 * (j & 3);                 // key index pair
        vf.u[j] = *reinterpret_cast<const unsigned int*>(&Vs[nc * 16 + lp][kk]);
      }
      oacc[nc] = wmma_f16(pf.v, vf.v, oacc[nc]);
    }
    __syncthreads();
  }

  // normalize and store f32
  #pragma unroll
  for (int nc = 0; nc < 4; ++nc)
    #pragma unroll
    for (int r = 0; r < 8; ++r) {
      const size_t row = (size_t)b * S_ + qb + w * 16 + r + 8 * hl;
      out[row * D_ + (size_t)h * DH_ + nc * 16 + lp] = oacc[nc][r] / l_r[r];
    }
}

// ---------------------------------------------------------------------------
extern "C" void kernel_launch(void* const* d_in, const int* in_sizes, int n_in,
                              void* d_out, int out_size, void* d_ws, size_t ws_size,
                              hipStream_t stream) {
  (void)in_sizes; (void)n_in; (void)out_size; (void)ws_size;
  const float* q  = (const float*)d_in[0];
  const float* v  = (const float*)d_in[1];
  const float* Wq = (const float*)d_in[2];
  const float* bq = (const float*)d_in[3];
  const float* Wk = (const float*)d_in[4];
  const float* bk = (const float*)d_in[5];
  const float* Wv = (const float*)d_in[6];
  const float* bv = (const float*)d_in[7];
  float* out = (float*)d_out;

  const size_t elems = (size_t)B_ * S_ * D_;     // 8M
  _Float16* qp16 = (_Float16*)d_ws;
  _Float16* kp16 = qp16 + elems;
  _Float16* vp16 = kp16 + elems;

  dim3 gG(B_ * S_ / 128, D_ / 128);              // (64, 8)
  proj_gemm_kernel<<<gG, 256, 0, stream>>>(q, Wq, bq, qp16);
  proj_gemm_kernel<<<gG, 256, 0, stream>>>(v, Wk, bk, kp16);
  proj_gemm_kernel<<<gG, 256, 0, stream>>>(v, Wv, bv, vp16);

  dim3 gA(S_ / 128, H_, B_);                     // (16, 16, 4)
  attn_kernel<<<gA, 256, 0, stream>>>(qp16, kp16, vp16, out);
}
